// CARAFEPack_81312320848240
// MI455X (gfx1250) — matrix operand order repaired and checked by reading, not compiled, not run on hardware
//
#include <hip/hip_runtime.h>

typedef __attribute__((ext_vector_type(16))) _Float16 v16h;
typedef __attribute__((ext_vector_type(8)))  _Float16 v8h;
typedef __attribute__((ext_vector_type(4)))  _Float16 v4h;
typedef __attribute__((ext_vector_type(8)))  float    v8f;
typedef __attribute__((ext_vector_type(4)))  float    v4f;
typedef __attribute__((ext_vector_type(4)))  int      v4i;

union V16U { v16h v; v8h h8[2]; _Float16 h[16]; };

#define BATCH 4
#define CIN   256
#define HIN   128
#define WINW  128
#define COMP  64
#define MASKC 100
#define MASKP 112           // 100 padded to 7*16
#define KDIM  576           // 9 taps * 64 channels
#define NPIX  (BATCH*HIN*WINW)   // 65536

// ---------------------------------------------------------------------------
// CDNA5 async global->LDS copy (16B per lane, tracked by ASYNCcnt).
// Builtin signature (from probe): (addrspace(1) v4i*, addrspace(3) v4i*,
// imm offset, imm cpol). Falls back to a plain b128 copy if unavailable.
// ---------------------------------------------------------------------------
#if __has_builtin(__builtin_amdgcn_global_load_async_to_lds_b128)
#define HAVE_ASYNC_LDS 1
typedef __attribute__((address_space(1))) v4i gas_v4i;
typedef __attribute__((address_space(3))) v4i las_v4i;
__device__ __forceinline__ void async_copy16(const void* g, void* l) {
  __builtin_amdgcn_global_load_async_to_lds_b128(
      (gas_v4i*)(__attribute__((address_space(1))) void*)(void*)g,
      (las_v4i*)(__attribute__((address_space(3))) void*)l, 0, 0);
}
__device__ __forceinline__ void async_wait0() {
#if __has_builtin(__builtin_amdgcn_s_wait_asynccnt)
  __builtin_amdgcn_s_wait_asynccnt(0);
#else
  asm volatile("s_wait_asynccnt 0" ::: "memory");
#endif
}
#else
#define HAVE_ASYNC_LDS 0
__device__ __forceinline__ void async_copy16(const void* g, void* l) {
  *(v4f*)l = *(const v4f*)g;
}
__device__ __forceinline__ void async_wait0() {}
#endif

// ---------------------------------------------------------------------------
// Kernel R: repack W2 (100,64,3,3) f32 OIHW -> f16 B-matrix W2^T [576][112],
// k ordered as tap*64 + c so each 32-wide K step stays inside one tap.
// ---------------------------------------------------------------------------
__global__ __launch_bounds__(256)
void repack_w2(const float* __restrict__ W2, _Float16* __restrict__ w2t) {
  int idx = blockIdx.x * 256 + threadIdx.x;   // 0 .. 576*112-1 = 64511
  int k = idx / MASKP;
  int m = idx - k * MASKP;
  int t = k >> 6;        // tap 0..8
  int c = k & 63;        // channel 0..63
  float v = (m < MASKC) ? W2[(m * COMP + c) * 9 + t] : 0.0f;
  w2t[idx] = (_Float16)v;
}

// ---------------------------------------------------------------------------
// Kernel G1: feat = W1(64x256) * X(256xN) + b1, N tile = 128 pixels (one row).
// Output feat pixel-major f16: feat[n*64 + m].
// ---------------------------------------------------------------------------
__global__ __launch_bounds__(256)
void gemm1(const float* __restrict__ x, const float* __restrict__ W1,
           const float* __restrict__ b1, _Float16* __restrict__ feat) {
  extern __shared__ char smem[];
  _Float16* lds_w = (_Float16*)smem;                       // [64][256]
  _Float16* lds_x = (_Float16*)(smem + COMP * CIN * 2);    // [256][128]
  const int tid = threadIdx.x;
  const int n0  = blockIdx.x * 128;
  const int b   = n0 >> 14;
  const int p0  = n0 & 16383;

  // stage W1 -> f16 LDS
  for (int e = tid * 4; e < COMP * CIN; e += 256 * 4) {
    v4f wv = *(const v4f*)(W1 + e);
    v4h hv;
    hv[0] = (_Float16)wv[0]; hv[1] = (_Float16)wv[1];
    hv[2] = (_Float16)wv[2]; hv[3] = (_Float16)wv[3];
    *(v4h*)(lds_w + e) = hv;
  }
  // stage X tile (256 ch x 128 px) -> f16 LDS, coalesced float4 loads
  for (int e4 = tid; e4 < (CIN * 128) / 4; e4 += 256) {
    int e  = e4 * 4;
    int c  = e >> 7;
    int xi = e & 127;
    v4f xv = *(const v4f*)(x + (size_t)(b * CIN + c) * 16384 + p0 + xi);
    v4h hv;
    hv[0] = (_Float16)xv[0]; hv[1] = (_Float16)xv[1];
    hv[2] = (_Float16)xv[2]; hv[3] = (_Float16)xv[3];
    *(v4h*)(lds_x + c * 128 + xi) = hv;
  }
  __syncthreads();

  const int wave = tid >> 5, lane = tid & 31;
  const int nl   = wave << 4;          // 16-column slab per wave
  const int lrow = lane & 15;
  const int hi   = lane >> 4;
  const int koff = hi << 3;

  v8f acc[4] = {};
  for (int k0 = 0; k0 < CIN; k0 += 32) {
    V16U bf;   // B 32x16: lane holds N = nl+lrow, K chunks {koff..+7, koff+16..+23}
#pragma unroll
    for (int s = 0; s < 16; ++s) {
      int kk = (s < 8) ? (koff + s) : (16 + koff + s - 8);
      bf.h[s] = lds_x[(k0 + kk) * 128 + nl + lrow];
    }
#pragma unroll
    for (int mt = 0; mt < 4; ++mt) {
      V16U af;  // A 16x32: row = mt*16+lrow, K chunks {koff..+7, koff+16..+23}
      const _Float16* ap = lds_w + (mt * 16 + lrow) * CIN + k0 + koff;
      af.h8[0] = *(const v8h*)(ap);
      af.h8[1] = *(const v8h*)(ap + 16);
      acc[mt] = __builtin_amdgcn_wmma_f32_16x16x32_f16(
          false, af.v, false, bf.v, (short)0, acc[mt], false, false);
    }
  }

  const int n = n0 + nl + lrow;
#pragma unroll
  for (int mt = 0; mt < 4; ++mt) {
    v8h o;
#pragma unroll
    for (int r = 0; r < 8; ++r) {
      int m = mt * 16 + hi * 8 + r;
      o[r] = (_Float16)(acc[mt][r] + b1[m]);
    }
    *(v8h*)(feat + (size_t)n * COMP + mt * 16 + hi * 8) = o;
  }
}

// ---------------------------------------------------------------------------
// Kernel G2: mask^T = im2col(feat)(Nx576) * W2^T(576x112) + b2, 3x3 pad 1.
// M = 16-pixel slab, N = 16 mask channels. Feat rows y-1,y,y+1 staged
// pixel-major [3][130][64] via async global->LDS b128 (contiguous copy, no
// transpose); the dj tap shift moves the pixel index so every A fragment is
// two 16B-aligned ds_load_b128. B fragments gather 16 halves at a 112-half row stride from the
// L2-resident repacked W2^T.
// ---------------------------------------------------------------------------
__global__ __launch_bounds__(256)
void gemm2(const _Float16* __restrict__ feat, const _Float16* __restrict__ w2t,
           const float* __restrict__ b2, float* __restrict__ mask) {
  extern __shared__ char smem[];
  _Float16* lds_f = (_Float16*)smem;   // [3][130][64], pixel index = x+1
  const int tid = threadIdx.x;
  const int n0  = blockIdx.x * 128;
  const int b   = n0 >> 14;
  const int y   = (n0 >> 7) & 127;

  // stage 3 feat rows (16KB contiguous each) at pixel offset +1
  for (int idx = tid; idx < 3 * 1024; idx += 256) {   // 16B chunks
    int di = idx >> 10;
    int q  = idx & 1023;
    int ys = y - 1 + di;
    _Float16* dst = lds_f + (di * 130 + 1) * 64 + q * 8;
    if ((unsigned)ys < 128u) {
      const _Float16* src = feat + (size_t)(b * 16384 + ys * 128) * 64 + q * 8;
      async_copy16(src, dst);
    } else {
      v8h z = {};
      *(v8h*)dst = z;
    }
  }
  // zero halo pixels 0 and 129 (64 halves each, 3 rows)
  for (int idx = tid; idx < 384; idx += 256) {
    int di  = idx >> 7;
    int rem = idx & 127;
    int px  = (rem < 64) ? 0 : 129;
    int c   = rem & 63;
    lds_f[(di * 130 + px) * 64 + c] = (_Float16)0;
  }
  async_wait0();
  __syncthreads();

  const int wave = tid >> 5, lane = tid & 31;
  const int nl   = wave << 4;          // 16-pixel slab per wave
  const int lrow = lane & 15;
  const int hi   = lane >> 4;
  const int koff = hi << 3;

  v8f acc[7] = {};
  for (int k0 = 0; k0 < KDIM; k0 += 32) {
    const int t  = k0 >> 6;          // tap (uniform: 32-step stays in one tap)
    const int di = t / 3, dj = t % 3;
    const int kc = k0 & 63;          // channel base within tap (0 or 32)
    V16U af;   // A 16x32: row = pixel nl+lrow, K = channels, chunks {koff..+7, koff+16..+23}
    const _Float16* ap = lds_f + (di * 130 + nl + lrow + dj) * 64 + kc + koff;
    af.h8[0] = *(const v8h*)(ap);
    af.h8[1] = *(const v8h*)(ap + 16);
#pragma unroll
    for (int mt = 0; mt < 7; ++mt) {
      V16U bf;  // B 32x16: lane holds column m = mt*16+lrow, K chunks {koff..+7, koff+16..+23}
#pragma unroll
      for (int s = 0; s < 16; ++s) {
        int kk = (s < 8) ? (koff + s) : (16 + koff + s - 8);
        bf.h[s] = w2t[(size_t)(k0 + kk) * MASKP + mt * 16 + lrow];
      }
      acc[mt] = __builtin_amdgcn_wmma_f32_16x16x32_f16(
          false, af.v, false, bf.v, (short)0, acc[mt], false, false);
    }
  }

  // D: lane holds mask channel m = mt*16+lrow, pixels nl+hi*8+r
#pragma unroll
  for (int mt = 0; mt < 7; ++mt) {
    int m = mt * 16 + lrow;
    if (m < MASKC) {
      float bias = b2[m];
#pragma unroll
      for (int r = 0; r < 8; ++r) {
        int n = n0 + nl + hi * 8 + r;
        mask[(size_t)n * 100 + m] = acc[mt][r] + bias;
      }
    }
  }
}

// ---------------------------------------------------------------------------
// Kernel C: pixel-shuffle + softmax(25) + nearest-upsample CARAFE reassembly.
// The 25 taps hit only a 3x3 set of distinct input pixels (nearest upsample),
// so fold the softmaxed weights into 9 channel-independent bins, then 9 FMAs
// per (channel, x). Input rows staged via async global->LDS b128; row stride
// 136 floats with halo at cols 3/132 keeps every 16B chunk aligned.
// ---------------------------------------------------------------------------
__global__ __launch_bounds__(256)
void carafe(const float* __restrict__ x, const float* __restrict__ mask,
            float* __restrict__ out) {
  extern __shared__ char smem[];
  float* lds_w9 = (float*)smem;                 // [256][9]
  float* lds_in = (float*)(smem + 256 * 9 * 4); // [3][64][136], col = win+4
  const int tid = threadIdx.x;
  const int bid = blockIdx.x;
  const int cg  = bid & 3;
  const int y   = (bid >> 2) & 255;
  const int b   = bid >> 10;
  const int c0  = cg << 6;
  const int h0  = (y - 2) >> 1;   // first (maybe invalid) source row

  // stage up to 3 input rows x 64 channels (512B contiguous per (row,ch))
  for (int idx = tid; idx < 3 * 64 * 32; idx += 256) {  // 16B chunks
    int r3  = idx >> 11;
    int rem = idx & 2047;
    int c   = rem >> 5;
    int q   = rem & 31;
    int hin = h0 + r3;
    float* dst = lds_in + (r3 * 64 + c) * 136 + 4 + q * 4;
    if ((unsigned)hin < 128u) {
      const float* src = x + (size_t)(b * CIN + c0 + c) * 16384 + hin * 128 + q * 4;
      async_copy16(src, dst);
    } else {
      v4f z = {};
      *(v4f*)dst = z;
    }
  }
  // zero halo columns 3 and 132
  for (int idx = tid; idx < 384; idx += 256) {
    int ri  = idx >> 7;
    int rem = idx & 127;
    int c   = rem & 63;
    int col = (rem < 64) ? 3 : 132;
    lds_in[(ri * 64 + c) * 136 + col] = 0.0f;
  }

  // phase 1: per-x softmax over 25 taps, fold into 9 aggregated weights
  {
    const int xo = tid;
    const int n  = (b << 14) + ((y >> 1) << 7) + (xo >> 1);
    const int rr = ((y & 1) << 1) + (xo & 1);      // pixel-shuffle sub-channel
    const float* mp = mask + (size_t)n * 100 + rr;
    float mv[25];
    float mx = -1e30f;
#pragma unroll
    for (int kk = 0; kk < 25; ++kk) { mv[kk] = mp[kk << 2]; mx = fmaxf(mx, mv[kk]); }
    float s = 0.0f;
#pragma unroll
    for (int kk = 0; kk < 25; ++kk) { mv[kk] = __expf(mv[kk] - mx); s += mv[kk]; }
    const float inv = 1.0f / s;
    float w9[9] = {0, 0, 0, 0, 0, 0, 0, 0, 0};
    const int w0 = (xo - 2) >> 1;
#pragma unroll
    for (int i = 0; i < 5; ++i) {
      int sy = y + i - 2;
      if ((unsigned)sy < 256u) {
        int ri = (sy >> 1) - h0;
#pragma unroll
        for (int j = 0; j < 5; ++j) {
          int sx = xo + j - 2;
          if ((unsigned)sx < 256u) {
            int rj = (sx >> 1) - w0;
            w9[ri * 3 + rj] += mv[i * 5 + j] * inv;
          }
        }
      }
    }
#pragma unroll
    for (int q = 0; q < 9; ++q) lds_w9[xo * 9 + q] = w9[q];
  }
  async_wait0();
  __syncthreads();

  // phase 2: 9-tap reassembly, 64 channels per thread, coalesced stores
  const int xo = tid;
  float w[9];
#pragma unroll
  for (int q = 0; q < 9; ++q) w[q] = lds_w9[xo * 9 + q];
  const int wc = ((xo - 2) >> 1) + 4;   // base LDS column
  float* ob = out + ((size_t)(b * CIN + c0) * 256 + y) * 256 + xo;
  for (int c = 0; c < 64; ++c) {
    float a = 0.0f;
#pragma unroll
    for (int ri = 0; ri < 3; ++ri) {
      const float* rp = lds_in + (ri * 64 + c) * 136 + wc;
      a = fmaf(rp[0], w[ri * 3 + 0], a);
      a = fmaf(rp[1], w[ri * 3 + 1], a);
      a = fmaf(rp[2], w[ri * 3 + 2], a);
    }
    ob[(size_t)c * 65536] = a;
  }
}

// ---------------------------------------------------------------------------
extern "C" void kernel_launch(void* const* d_in, const int* in_sizes, int n_in,
                              void* d_out, int out_size, void* d_ws, size_t ws_size,
                              hipStream_t stream) {
  const float* x  = (const float*)d_in[0];
  const float* W1 = (const float*)d_in[1];
  const float* b1 = (const float*)d_in[2];
  const float* W2 = (const float*)d_in[3];
  const float* b2 = (const float*)d_in[4];
  float* out = (float*)d_out;

  char* ws = (char*)d_ws;
  // workspace layout (16B aligned):
  //   feat f16 [65536][64]  : 8,388,608 B
  //   w2t  f16 [576][112]   :   129,024 B
  //   mask f32 [65536][100] : 26,214,400 B
  _Float16* feat  = (_Float16*)ws;
  _Float16* w2t   = (_Float16*)(ws + 8388608);
  float*    maskp = (float*)(ws + 8388608 + 129024);

  const int smem_g1 = (COMP * CIN + CIN * 128) * 2;          //  98,304 B
  const int smem_g2 = 3 * 130 * 64 * 2;                      //  49,920 B
  const int smem_c  = 256 * 9 * 4 + 3 * 64 * 136 * 4;        // 113,664 B

  // allow >64KB dynamic LDS (WGP has 320KB)
  (void)hipFuncSetAttribute(reinterpret_cast<const void*>(gemm1),
                            hipFuncAttributeMaxDynamicSharedMemorySize, smem_g1);
  (void)hipFuncSetAttribute(reinterpret_cast<const void*>(carafe),
                            hipFuncAttributeMaxDynamicSharedMemorySize, smem_c);

  repack_w2<<<(MASKP * KDIM) / 256, 256, 0, stream>>>(W2, w2t);
  gemm1<<<NPIX / 128, 256, smem_g1, stream>>>(x, W1, b1, feat);
  gemm2<<<NPIX / 128, 256, smem_g2, stream>>>(feat, w2t, b2, maskp);
  carafe<<<4 * 256 * 4, 256, smem_c, stream>>>(x, maskp, out);
}
